// WSIEncoder_12867722019336
// MI455X (gfx1250) — compile-verified, matrix operand order checked by reading, not run on hardware
//
#include <hip/hip_runtime.h>
#include <hip/hip_bf16.h>
#include <stdint.h>

typedef __attribute__((ext_vector_type(16))) _Float16 v16h;
typedef __attribute__((ext_vector_type(8)))  _Float16 h8;
typedef __attribute__((ext_vector_type(8)))  float    v8f;
typedef __attribute__((ext_vector_type(2)))  float    v2f;

#define BATCH  4
#define SEQ    4096
#define DMODEL 512
#define NHEADS 8
#define DH     64
#define LMK    256
#define BH     (BATCH*NHEADS)

// ---------------------------------------------------------------------------
// f16 NT GEMM: C(MxN) = A(MxK) * B(NxK)^T, f32 accum via v_wmma_f32_16x16x32_f16
// Block tile BM x BN, wave tile 32x64, K step 32. lda=ldb=K, ldc=N.
// Double-buffered LDS (register staging) + global_prefetch of tile k+2.
// ---------------------------------------------------------------------------
template<int BM, int BN>
__global__ __launch_bounds__((BM/32)*(BN/64)*32)
void gemm_nt_f16(const _Float16* __restrict__ A, const _Float16* __restrict__ B,
                 float* __restrict__ Cf, _Float16* __restrict__ Ch,
                 int M, int N, int K, long sA_, long sB_, long sC_)
{
    constexpr int KB = 32;
    constexpr int WAVES_M = BM / 32;
    constexpr int WAVES_N = BN / 64;
    constexpr int THREADS = WAVES_M * WAVES_N * 32;
    constexpr int LDR = KB + 8;                 // 40 halves = 80B rows, 16B aligned
    constexpr int NLA = (BM * 4) / THREADS;     // 8-half chunks per thread (A)
    constexpr int NLB = (BN * 4) / THREADS;     // 8-half chunks per thread (B)
    __shared__ _Float16 sA[2][BM * LDR];
    __shared__ _Float16 sB[2][BN * LDR];

    const int tid  = threadIdx.x;
    const int wave = tid >> 5, lane = tid & 31;
    const int g  = lane >> 4, rr = lane & 15;
    const int wm = wave / WAVES_N, wn = wave % WAVES_N;
    const int bm0 = blockIdx.y * BM, bn0 = blockIdx.x * BN;
    const long batch = blockIdx.z;
    const _Float16* Ab = A + batch * sA_;
    const _Float16* Bb = B + batch * sB_;

    v8f acc[2][4];
    v8f zero = {0.f,0.f,0.f,0.f,0.f,0.f,0.f,0.f};
    #pragma unroll
    for (int i = 0; i < 2; ++i)
        #pragma unroll
        for (int j = 0; j < 4; ++j) acc[i][j] = zero;

    union Frag { v16h v; h8 h[2]; };
    h8 ra[NLA], rb[NLB];

    // prologue: fetch tile 0 into registers, commit to LDS buffer 0
    #pragma unroll
    for (int j = 0; j < NLA; ++j) {
        int i = tid + j * THREADS;
        int r = i >> 2, c = (i & 3) << 3;
        ra[j] = *(const h8*)&Ab[(long)(bm0 + r) * K + c];
    }
    #pragma unroll
    for (int j = 0; j < NLB; ++j) {
        int i = tid + j * THREADS;
        int r = i >> 2, c = (i & 3) << 3;
        rb[j] = *(const h8*)&Bb[(long)(bn0 + r) * K + c];
    }
    #pragma unroll
    for (int j = 0; j < NLA; ++j) {
        int i = tid + j * THREADS;
        int r = i >> 2, c = (i & 3) << 3;
        *(h8*)&sA[0][r * LDR + c] = ra[j];
    }
    #pragma unroll
    for (int j = 0; j < NLB; ++j) {
        int i = tid + j * THREADS;
        int r = i >> 2, c = (i & 3) << 3;
        *(h8*)&sB[0][r * LDR + c] = rb[j];
    }
    __syncthreads();

    int buf = 0;
    for (int k0 = 0; k0 < K; k0 += KB) {
        const bool hasNext = (k0 + KB) < K;

        // prefetch tile k+2 toward L2/L0 (global_prefetch_b8)
        if (k0 + 2 * KB < K) {
            __builtin_prefetch(&Ab[(long)(bm0 + (tid >> 2)) * K + k0 + 2 * KB], 0, 3);
            __builtin_prefetch(&Bb[(long)(bn0 + (tid & (BN - 1))) * K + k0 + 2 * KB], 0, 3);
        }

        // stage tile k+1 into registers while WMMAs consume tile k from LDS
        if (hasNext) {
            #pragma unroll
            for (int j = 0; j < NLA; ++j) {
                int i = tid + j * THREADS;
                int r = i >> 2, c = (i & 3) << 3;
                ra[j] = *(const h8*)&Ab[(long)(bm0 + r) * K + k0 + KB + c];
            }
            #pragma unroll
            for (int j = 0; j < NLB; ++j) {
                int i = tid + j * THREADS;
                int r = i >> 2, c = (i & 3) << 3;
                rb[j] = *(const h8*)&Bb[(long)(bn0 + r) * K + k0 + KB + c];
            }
        }

        // fragments from LDS (documented 16-bit A layout; B is NxK -> same loader)
        Frag a[2], b[4];
        #pragma unroll
        for (int mi = 0; mi < 2; ++mi) {
            int row = (wm * 32 + mi * 16 + rr) * LDR;
            a[mi].h[0] = *(const h8*)&sA[buf][row + 8 * g];       // K = 8g..8g+7
            a[mi].h[1] = *(const h8*)&sA[buf][row + 16 + 8 * g];  // K = 16+8g..
        }
        #pragma unroll
        for (int ni = 0; ni < 4; ++ni) {
            int row = (wn * 64 + ni * 16 + rr) * LDR;
            b[ni].h[0] = *(const h8*)&sB[buf][row + 8 * g];
            b[ni].h[1] = *(const h8*)&sB[buf][row + 16 + 8 * g];
        }
        #pragma unroll
        for (int mi = 0; mi < 2; ++mi)
            #pragma unroll
            for (int ni = 0; ni < 4; ++ni)
                acc[mi][ni] = __builtin_amdgcn_wmma_f32_16x16x32_f16(
                    false, a[mi].v, false, b[ni].v, (short)0, acc[mi][ni], false, false);

        // commit tile k+1 into the other buffer; one barrier per K-step
        if (hasNext) {
            #pragma unroll
            for (int j = 0; j < NLA; ++j) {
                int i = tid + j * THREADS;
                int r = i >> 2, c = (i & 3) << 3;
                *(h8*)&sA[buf ^ 1][r * LDR + c] = ra[j];
            }
            #pragma unroll
            for (int j = 0; j < NLB; ++j) {
                int i = tid + j * THREADS;
                int r = i >> 2, c = (i & 3) << 3;
                *(h8*)&sB[buf ^ 1][r * LDR + c] = rb[j];
            }
            __syncthreads();
            buf ^= 1;
        }
    }

    #pragma unroll
    for (int mi = 0; mi < 2; ++mi)
        #pragma unroll
        for (int ni = 0; ni < 4; ++ni) {
            long crow = bm0 + wm * 32 + mi * 16 + 8 * g;       // M = vgpr + 8*(lane/16)
            long ccol = bn0 + wn * 64 + ni * 16 + rr;          // N = lane%16
            long cb = batch * sC_ + crow * (long)N + ccol;
            #pragma unroll
            for (int i = 0; i < 8; ++i) {
                float v = acc[mi][ni][i];
                if (Cf) Cf[cb + (long)i * N] = v;
                if (Ch) Ch[cb + (long)i * N] = (_Float16)v;
            }
        }
}

// ---------------------------------------------------------------------------
// f32 NN GEMM 256x256x256 batched (pinv iteration) via v_wmma_f32_16x16x4_f32
// ---------------------------------------------------------------------------
__global__ __launch_bounds__(256)
void gemm_nn_f32_256(const float* __restrict__ A, const float* __restrict__ B,
                     float* __restrict__ C, float alpha)
{
    const long bofs = (long)blockIdx.z << 16;
    const float* Ab = A + bofs;
    const float* Bb = B + bofs;
    int tid = threadIdx.x, wave = tid >> 5, lane = tid & 31;
    int g = lane >> 4, rr = lane & 15;
    int mt = blockIdx.x * 8 + wave;     // 16 row-tiles of 16
    int n0 = blockIdx.y * 64;
    int m  = mt * 16 + rr;

    v8f acc[4];
    v8f zero = {0.f,0.f,0.f,0.f,0.f,0.f,0.f,0.f};
    #pragma unroll
    for (int i = 0; i < 4; ++i) acc[i] = zero;

    for (int k0 = 0; k0 < 256; k0 += 4) {
        int k = k0 + 2 * g;
        v2f a;
        a[0] = Ab[(long)m * 256 + k];
        a[1] = Ab[(long)m * 256 + k + 1];
        #pragma unroll
        for (int ni = 0; ni < 4; ++ni) {
            int col = n0 + ni * 16 + rr;
            v2f b;
            b[0] = Bb[(long)k * 256 + col];
            b[1] = Bb[(long)(k + 1) * 256 + col];
            acc[ni] = __builtin_amdgcn_wmma_f32_16x16x4_f32(
                false, a, false, b, (short)0, acc[ni], false, false);
        }
    }
    #pragma unroll
    for (int ni = 0; ni < 4; ++ni) {
        long row0 = mt * 16 + 8 * g;
        long col  = n0 + ni * 16 + rr;
        #pragma unroll
        for (int i = 0; i < 8; ++i)
            C[bofs + (row0 + i) * 256 + col] = alpha * acc[ni][i];
    }
}

// W = c*I - X   (batched 32 x 256x256)
__global__ __launch_bounds__(256)
void diag_cmi(const float* __restrict__ X, float* __restrict__ W, float c)
{
    long idx = (long)blockIdx.x * 256 + threadIdx.x;
    int ij = (int)(idx & 65535);
    int i = ij >> 8, j = ij & 255;
    W[idx] = ((i == j) ? c : 0.f) - X[idx];
}

// per-batch max |row-sum| and max |col-sum|
__global__ __launch_bounds__(256)
void pinv_absmax(const float* __restrict__ X, float* __restrict__ outp)
{
    int b = blockIdx.x;
    const float* P = X + ((long)b << 16);
    int t = threadIdx.x;
    float rs = 0.f, cs = 0.f;
    for (int j = 0; j < 256; ++j) {
        rs += fabsf(P[t * 256 + j]);
        cs += fabsf(P[j * 256 + t]);
    }
    __shared__ float sm[512];
    sm[t] = rs; sm[256 + t] = cs;
    __syncthreads();
    for (int s = 128; s > 0; s >>= 1) {
        if (t < s) {
            sm[t]       = fmaxf(sm[t], sm[t + s]);
            sm[256 + t] = fmaxf(sm[256 + t], sm[256 + t + s]);
        }
        __syncthreads();
    }
    if (t == 0) { outp[b * 2] = sm[0]; outp[b * 2 + 1] = sm[256]; }
}

__global__ __launch_bounds__(32)
void pinv_scale(const float* __restrict__ mx, float* __restrict__ scale)
{
    if (threadIdx.x == 0) {
        float ma = 0.f, mb = 0.f;
        for (int i = 0; i < BH; ++i) {
            ma = fmaxf(ma, mx[i * 2]);
            mb = fmaxf(mb, mx[i * 2 + 1]);
        }
        scale[0] = 1.f / (ma * mb);
    }
}

__global__ __launch_bounds__(256)
void pinv_tscale(const float* __restrict__ X, const float* __restrict__ scale,
                 float* __restrict__ Z)
{
    long idx = (long)blockIdx.x * 256 + threadIdx.x;
    long b = idx >> 16;
    int ij = (int)(idx & 65535);
    int i = ij >> 8, j = ij & 255;
    Z[(b << 16) + (i << 8) + j] = X[(b << 16) + (j << 8) + i] * scale[0];
}

// ---------------------------------------------------------------------------
// softmax kernels
// ---------------------------------------------------------------------------
__global__ __launch_bounds__(256)
void softmax256(const float* __restrict__ in, _Float16* __restrict__ oh,
                float* __restrict__ of)
{
    long row = (long)blockIdx.x * 8 + (threadIdx.x >> 5);
    int lane = threadIdx.x & 31;
    const float* p = in + row * 256;
    float v[8], mx = -1e30f;
    #pragma unroll
    for (int j = 0; j < 8; ++j) { v[j] = p[lane + j * 32]; mx = fmaxf(mx, v[j]); }
    for (int s = 16; s > 0; s >>= 1) mx = fmaxf(mx, __shfl_xor(mx, s, 32));
    float sum = 0.f;
    #pragma unroll
    for (int j = 0; j < 8; ++j) { v[j] = __expf(v[j] - mx); sum += v[j]; }
    for (int s = 16; s > 0; s >>= 1) sum += __shfl_xor(sum, s, 32);
    float r = 1.f / sum;
    #pragma unroll
    for (int j = 0; j < 8; ++j) {
        float o = v[j] * r;
        long idx = row * 256 + lane + j * 32;
        if (oh) oh[idx] = (_Float16)o;
        if (of) of[idx] = o;
    }
}

__global__ __launch_bounds__(256)
void softmax4096(const float* __restrict__ in, _Float16* __restrict__ oh)
{
    long row = blockIdx.x;
    const float* p = in + row * 4096;
    int t = threadIdx.x;
    float v[16], mx = -1e30f;
    #pragma unroll
    for (int j = 0; j < 16; ++j) { v[j] = p[t + j * 256]; mx = fmaxf(mx, v[j]); }
    __shared__ float sm[256];
    sm[t] = mx; __syncthreads();
    for (int s = 128; s > 0; s >>= 1) { if (t < s) sm[t] = fmaxf(sm[t], sm[t + s]); __syncthreads(); }
    mx = sm[0]; __syncthreads();
    float sum = 0.f;
    #pragma unroll
    for (int j = 0; j < 16; ++j) { v[j] = __expf(v[j] - mx); sum += v[j]; }
    sm[t] = sum; __syncthreads();
    for (int s = 128; s > 0; s >>= 1) { if (t < s) sm[t] += sm[t + s]; __syncthreads(); }
    float r = 1.f / sm[0];
    #pragma unroll
    for (int j = 0; j < 16; ++j)
        oh[row * 4096 + t + j * 256] = (_Float16)(v[j] * r);
}

// ---------------------------------------------------------------------------
// layernorm -> f16 (wave per 512-wide row)
// ---------------------------------------------------------------------------
__global__ __launch_bounds__(256)
void layernorm_f16(const float* __restrict__ x, const float* __restrict__ gamma,
                   const float* __restrict__ beta, _Float16* __restrict__ xn)
{
    long row = (long)blockIdx.x * 8 + (threadIdx.x >> 5);
    int lane = threadIdx.x & 31;
    const float* p = x + row * DMODEL;
    float v[16], s = 0.f;
    #pragma unroll
    for (int j = 0; j < 16; ++j) { v[j] = p[lane + j * 32]; s += v[j]; }
    for (int o = 16; o > 0; o >>= 1) s += __shfl_xor(s, o, 32);
    float mu = s * (1.f / DMODEL);
    float var = 0.f;
    #pragma unroll
    for (int j = 0; j < 16; ++j) { float d = v[j] - mu; var += d * d; }
    for (int o = 16; o > 0; o >>= 1) var += __shfl_xor(var, o, 32);
    float inv = rsqrtf(var * (1.f / DMODEL) + 1e-5f);
    #pragma unroll
    for (int j = 0; j < 16; ++j) {
        int c = lane + j * 32;
        xn[row * DMODEL + c] = (_Float16)((v[j] - mu) * inv * gamma[c] + beta[c]);
    }
}

// transpose + convert f32 -> f16, batched: in (z,R,C) -> out (z,C,R)
__global__ __launch_bounds__(256)
void tconv_f32_f16(const float* __restrict__ in, _Float16* __restrict__ outp,
                   int R, int C)
{
    long base = (long)blockIdx.y * R * C;
    int idx = blockIdx.x * 256 + threadIdx.x;
    if (idx < R * C) {
        int r = idx / C, c = idx % C;
        outp[base + (long)c * R + r] = (_Float16)in[base + idx];
    }
}

// split qkv f32 (B*N,1536) -> qh(f16,scaled), kh(f16), vT(f16, DHxN), vf(f32)
__global__ __launch_bounds__(256)
void split_qkv(const float* __restrict__ qkv, _Float16* __restrict__ qh,
               _Float16* __restrict__ kh, _Float16* __restrict__ vT,
               float* __restrict__ vf)
{
    long idx = (long)blockIdx.x * 256 + threadIdx.x;  // B*N*512
    long bn = idx >> 9;
    int c = (int)(idx & 511);
    int h = c >> 6, d = c & 63;
    long b = bn >> 12, n = bn & 4095;
    const float* row = qkv + bn * (3 * DMODEL);
    float q = row[h * 64 + d];
    float k = row[DMODEL + h * 64 + d];
    float v = row[2 * DMODEL + h * 64 + d];
    long o = (((b * NHEADS + h) * SEQ) + n) * DH + d;
    qh[o] = (_Float16)(q * 0.125f);   // DH^-0.5
    kh[o] = (_Float16)k;
    vf[o] = v;
    vT[(((b * NHEADS + h) * DH) + d) * (long)SEQ + n] = (_Float16)v;
}

// landmark means over groups of 16 tokens
__global__ __launch_bounds__(256)
void landmark_means(const _Float16* __restrict__ qh, const _Float16* __restrict__ kh,
                    _Float16* __restrict__ ql, _Float16* __restrict__ kl)
{
    long idx = (long)blockIdx.x * 256 + threadIdx.x;  // BH*256*64
    long bh = idx / (LMK * DH);
    int rem = (int)(idx % (LMK * DH));
    int mi = rem / DH, d = rem % DH;
    const _Float16* pq = qh + (bh * SEQ + (long)mi * 16) * DH + d;
    const _Float16* pk = kh + (bh * SEQ + (long)mi * 16) * DH + d;
    float sq = 0.f, sk = 0.f;
    #pragma unroll
    for (int l = 0; l < 16; ++l) {
        sq += (float)pq[l * DH];
        sk += (float)pk[l * DH];
    }
    ql[idx] = (_Float16)(sq * (1.f / 16.f));
    kl[idx] = (_Float16)(sk * (1.f / 16.f));
}

// depthwise conv residual on v + repack heads -> O f16 (B,N,H*DH)
__global__ __launch_bounds__(64)
void conv_pack(const float* __restrict__ oc, const float* __restrict__ vf,
               const float* __restrict__ w, _Float16* __restrict__ O)
{
    int n = blockIdx.x, h = blockIdx.y, b = blockIdx.z, d = threadIdx.x;
    long bh = (long)b * NHEADS + h;
    float acc = oc[(bh * SEQ + n) * DH + d];
    const float* wk = w + h * 33;
    #pragma unroll
    for (int k = 0; k < 33; ++k) {
        int nn = n + k - 16;
        if (nn >= 0 && nn < SEQ) acc += wk[k] * vf[(bh * SEQ + nn) * DH + d];
    }
    O[((long)b * SEQ + n) * DMODEL + h * DH + d] = (_Float16)acc;
}

// out = x + y0 + b_out
__global__ __launch_bounds__(256)
void final_add(const float* __restrict__ x, const float* __restrict__ y0,
               const float* __restrict__ b, float* __restrict__ outp)
{
    long idx = (long)blockIdx.x * 256 + threadIdx.x;
    outp[idx] = x[idx] + y0[idx] + b[idx & (DMODEL - 1)];
}

// ---------------------------------------------------------------------------
extern "C" void kernel_launch(void* const* d_in, const int* in_sizes, int n_in,
                              void* d_out, int out_size, void* d_ws, size_t ws_size,
                              hipStream_t stream)
{
    (void)in_sizes; (void)n_in; (void)out_size; (void)ws_size;
    const float* x     = (const float*)d_in[0];
    const float* gamma = (const float*)d_in[1];
    const float* beta  = (const float*)d_in[2];
    const float* Wqkv  = (const float*)d_in[3];
    const float* Wout  = (const float*)d_in[4];
    const float* bout  = (const float*)d_in[5];
    const float* convw = (const float*)d_in[6];
    float* out = (float*)d_out;

    char* base = (char*)d_ws;
    size_t off = 0;
    auto alloc = [&](size_t bytes) -> char* {
        off = (off + 255) & ~(size_t)255;
        char* p = base + off; off += bytes; return p;
    };
    _Float16* xn    = (_Float16*)alloc(8388608ull * 2);   // (16384,512) f16
    _Float16* wqT   = (_Float16*)alloc(786432ull * 2);    // Wqkv^T f16
    _Float16* woT   = (_Float16*)alloc(262144ull * 2);    // Wout^T f16
    float*    bigf  = (float*)   alloc(33554432ull * 4);  // qkv f32 / sim1 / sim3
    _Float16* qh    = (_Float16*)alloc(8388608ull * 2);   // (BH,N,64) scaled
    _Float16* kh    = (_Float16*)alloc(8388608ull * 2);
    _Float16* vT    = (_Float16*)alloc(8388608ull * 2);   // (BH,64,N)
    float*    vf    = (float*)   alloc(8388608ull * 4);
    _Float16* ql    = (_Float16*)alloc(524288ull * 2);    // (BH,256,64)
    _Float16* kl    = (_Float16*)alloc(524288ull * 2);
    _Float16* attn1 = (_Float16*)alloc(33554432ull * 2);  // (BH,N,256)
    _Float16* attn3 = (_Float16*)alloc(33554432ull * 2);  // (BH,256,N) -> reused as tmp
    float*    attn2 = (float*)   alloc(2097152ull * 4);   // (BH,256,256)
    float*    z0    = (float*)   alloc(2097152ull * 4);
    float*    z1    = (float*)   alloc(2097152ull * 4);
    float*    xzb   = (float*)   alloc(2097152ull * 4);
    float*    wa    = (float*)   alloc(2097152ull * 4);
    float*    wb    = (float*)   alloc(2097152ull * 4);
    _Float16* zt    = (_Float16*)alloc(2097152ull * 2);   // Z^T f16
    float*    rf    = (float*)   alloc(524288ull * 4);    // (BH,256,64)
    _Float16* rT    = (_Float16*)alloc(524288ull * 2);    // (BH,64,256)
    float*    ocf   = (float*)   alloc(8388608ull * 4);   // out_core / Y0
    _Float16* Oh    = (_Float16*)alloc(8388608ull * 2);   // (B,N,512) f16
    float*    mx    = (float*)   alloc(65 * 4);
    float*    scl   = mx + 64;

    // weight transposes -> f16
    tconv_f32_f16<<<dim3(3072, 1), 256, 0, stream>>>(Wqkv, wqT, 512, 1536);
    tconv_f32_f16<<<dim3(1024, 1), 256, 0, stream>>>(Wout, woT, 512, 512);

    // layernorm
    layernorm_f16<<<2048, 256, 0, stream>>>(x, gamma, beta, xn);

    // QKV GEMM: (16384x512) x (512x1536)
    gemm_nt_f16<128, 128><<<dim3(12, 128, 1), 256, 0, stream>>>(
        xn, wqT, bigf, nullptr, 16384, 1536, 512, 0, 0, 0);

    split_qkv<<<32768, 256, 0, stream>>>(bigf, qh, kh, vT, vf);
    landmark_means<<<2048, 256, 0, stream>>>(qh, kh, ql, kl);

    // sim1 = q @ k_l^T : (N x 256), K=64, batched BH
    gemm_nt_f16<128, 128><<<dim3(2, 32, BH), 256, 0, stream>>>(
        qh, kl, bigf, nullptr, SEQ, LMK, DH,
        (long)SEQ * DH, (long)LMK * DH, (long)SEQ * LMK);
    softmax256<<<16384, 256, 0, stream>>>(bigf, attn1, nullptr);

    // sim2 = q_l @ k_l^T : (256 x 256), K=64
    gemm_nt_f16<128, 128><<<dim3(2, 2, BH), 256, 0, stream>>>(
        ql, kl, attn2, nullptr, LMK, LMK, DH,
        (long)LMK * DH, (long)LMK * DH, (long)LMK * LMK);
    softmax256<<<1024, 256, 0, stream>>>(attn2, nullptr, attn2);  // in-place f32

    // Moore-Penrose pinv init: z = x^T / (max col-sum * max row-sum), global max
    pinv_absmax<<<BH, 256, 0, stream>>>(attn2, mx);
    pinv_scale<<<1, 32, 0, stream>>>(mx, scl);
    pinv_tscale<<<8192, 256, 0, stream>>>(attn2, scl, z0);

    float* zc = z0; float* zn = z1;
    for (int it = 0; it < 6; ++it) {
        gemm_nn_f32_256<<<dim3(2, 4, BH), 256, 0, stream>>>(attn2, zc, xzb, 1.f);
        diag_cmi<<<8192, 256, 0, stream>>>(xzb, wa, 7.f);
        gemm_nn_f32_256<<<dim3(2, 4, BH), 256, 0, stream>>>(xzb, wa, wb, 1.f);
        diag_cmi<<<8192, 256, 0, stream>>>(wb, wa, 15.f);
        gemm_nn_f32_256<<<dim3(2, 4, BH), 256, 0, stream>>>(xzb, wa, wb, 1.f);
        diag_cmi<<<8192, 256, 0, stream>>>(wb, wa, 13.f);
        gemm_nn_f32_256<<<dim3(2, 4, BH), 256, 0, stream>>>(zc, wa, zn, 0.25f);
        float* t = zc; zc = zn; zn = t;
    }
    // final Z in zc; Z^T -> f16 for NT GEMM
    tconv_f32_f16<<<dim3(256, BH), 256, 0, stream>>>(zc, zt, 256, 256);

    // sim3 = q_l @ k^T : (256 x N), K=64
    gemm_nt_f16<128, 128><<<dim3(32, 2, BH), 256, 0, stream>>>(
        ql, kh, bigf, nullptr, LMK, SEQ, DH,
        (long)LMK * DH, (long)SEQ * DH, (long)LMK * SEQ);
    softmax4096<<<8192, 256, 0, stream>>>(bigf, attn3);

    // r = attn3 @ v : (256 x 64), K=N
    gemm_nt_f16<128, 64><<<dim3(1, 2, BH), 128, 0, stream>>>(
        attn3, vT, rf, nullptr, LMK, DH, SEQ,
        (long)LMK * SEQ, (long)DH * SEQ, (long)LMK * DH);
    tconv_f32_f16<<<dim3(64, BH), 256, 0, stream>>>(rf, rT, 256, 64);

    // tmp = attn1 @ Z : (N x 256), K=256  -> f16, reuse attn3 buffer
    _Float16* tmp = attn3;
    gemm_nt_f16<128, 128><<<dim3(2, 32, BH), 256, 0, stream>>>(
        attn1, zt, nullptr, tmp, SEQ, LMK, LMK,
        (long)SEQ * LMK, (long)LMK * LMK, (long)SEQ * LMK);

    // out_core = tmp @ r : (N x 64), K=256
    gemm_nt_f16<128, 64><<<dim3(1, 32, BH), 128, 0, stream>>>(
        tmp, rT, ocf, nullptr, SEQ, DH, LMK,
        (long)SEQ * LMK, (long)DH * LMK, (long)SEQ * DH);

    // + depthwise conv residual on v, repack heads -> O f16
    conv_pack<<<dim3(SEQ, NHEADS, BATCH), 64, 0, stream>>>(ocf, vf, convw, Oh);

    // Y0 = O @ W_out : (16384x512), K=512  (reuse ocf)
    gemm_nt_f16<128, 128><<<dim3(4, 128, 1), 256, 0, stream>>>(
        Oh, woT, ocf, nullptr, 16384, 512, 512, 0, 0, 0);

    // out = x + Y0 + b_out
    final_add<<<32768, 256, 0, stream>>>(x, ocf, bout, out);
}
